// SimpleModel_77412490543860
// MI455X (gfx1250) — matrix operand order repaired and checked by reading, not compiled, not run on hardware
//
#include <hip/hip_runtime.h>
#include <hip/hip_bf16.h>

// ---------------------------------------------------------------------------
// MI455X (gfx1250) implementation.
//
// Dead-code analysis of the reference (exact, not approximate):
//  * a0 MHA has Lk==1  -> softmax over singleton == 1.0 exactly -> output is
//    v0 broadcast; board/conv/PE path and a0 q/k projections are dead.
//  * state therefore has 200 identical rows; all (B*SEQ)xK GEMMs collapse to
//    BxK GEMMs.
//  * a1 MHA keys are 200 identical rows -> uniform softmax -> output == v2;
//    current/xyrot/loc1/loc2 and a1 q/k projections are dead.
// Live work: ~12 GEMMs of (512|2048) x (512|2048) x 512  ==> WMMA f16/f32acc.
//
// Round-2 changes:
//  * LDS-tiled weight transpose (coalesced on both sides) — the naive version
//    had stride-2KB reads (~32x amplification on ~15MB of weights, which
//    would have dominated the runtime).
//  * GEMM epilogue now writes optional f32 and/or f16 outputs directly from
//    the accumulators — removes all standalone f32->f16 passes.
// ---------------------------------------------------------------------------

typedef __attribute__((ext_vector_type(16))) _Float16 v16h;
typedef __attribute__((ext_vector_type(8)))  _Float16 v8h;
typedef __attribute__((ext_vector_type(8)))  float    v8f;

#define BATCH 512
#define DMODEL 512

static inline unsigned cdivu(unsigned a, unsigned b) { return (a + b - 1) / b; }

// ---------------------------------------------------------------------------
// WMMA GEMM:  D[M,N] = A16[M,K] @ Bt16[K,N] + bias[N]
// f32 out to C (if non-null) and f16 out to C16 (if non-null).
// wave tile 32x64, block = 4 waves -> 64x128 block tile. M%64==0, N%128==0, K%32==0.
// ---------------------------------------------------------------------------
__global__ __launch_bounds__(128) void wmma_gemm_bias_kernel(
    const _Float16* __restrict__ A, const _Float16* __restrict__ Bt,
    const float* __restrict__ bias,
    float* __restrict__ C, _Float16* __restrict__ C16,
    int M, int N, int K)
{
  const int lane = threadIdx.x & 31;
  const int wave = threadIdx.x >> 5;
  const int wm = wave >> 1, wn = wave & 1;
  const int tileM = blockIdx.y * 64 + wm * 32;
  const int tileN = blockIdx.x * 128 + wn * 64;
  const int half = lane >> 4;   // which 16-lane group
  const int l16  = lane & 15;

  // C/D layout: VGPR r of acc -> row (r + 8*half), col l16  (16x16 f32 tile).
  v8f acc[2][4];
#pragma unroll
  for (int ni = 0; ni < 4; ++ni) {
    float bv = bias[tileN + ni * 16 + l16];
#pragma unroll
    for (int mi = 0; mi < 2; ++mi)
#pragma unroll
      for (int r = 0; r < 8; ++r) acc[mi][ni][r] = bv;
  }

  const int kk = l16 + 16 * half;   // next-step Bt row (0..31) this lane prefetches

  for (int k0 = 0; k0 < K; k0 += 32) {
    // A fragment (16x32 f16): lane (l16, half) holds row l16,
    // K = k0 + [8*half .. 8*half+7] and k0 + [16+8*half .. 23+8*half].
    v16h a[2], b[4];
#pragma unroll
    for (int mi = 0; mi < 2; ++mi) {
      const _Float16* ap = A + (size_t)(tileM + mi * 16 + l16) * K + k0 + 8 * half;
      v8h lo = *(const v8h*)ap;
      v8h hi = *(const v8h*)(ap + 16);
      a[mi] = __builtin_shufflevector(lo, hi, 0,1,2,3,4,5,6,7,8,9,10,11,12,13,14,15);
    }
    // B fragment (32x16 f16): lane holds column l16, K = k0 + [8*half .. 8*half+7] and k0 + [16+8*half .. 23+8*half].
    const _Float16* bp = Bt + (size_t)k0 * N + tileN + l16;
#pragma unroll
    for (int ni = 0; ni < 4; ++ni) {
#pragma unroll
      for (int s = 0; s < 16; ++s) {
        int kr = (s < 8) ? (8 * half + s) : (16 + 8 * half + (s - 8));
        b[ni][s] = bp[(size_t)kr * N + ni * 16];
      }
    }
    if (k0 + 32 < K) {            // speculative prefetch of next K-step
      __builtin_prefetch(A + (size_t)(tileM + l16) * K + k0 + 32, 0, 1);
      __builtin_prefetch(Bt + (size_t)(k0 + 32 + kk) * N + tileN, 0, 1);
    }
#pragma unroll
    for (int mi = 0; mi < 2; ++mi)
#pragma unroll
      for (int ni = 0; ni < 4; ++ni)
        acc[mi][ni] = __builtin_amdgcn_wmma_f32_16x16x32_f16(
            false, a[mi], false, b[ni], (short)0, acc[mi][ni], false, false);
  }

#pragma unroll
  for (int mi = 0; mi < 2; ++mi)
#pragma unroll
    for (int ni = 0; ni < 4; ++ni)
#pragma unroll
      for (int r = 0; r < 8; ++r) {
        int m = tileM + mi * 16 + r + 8 * half;
        int n = tileN + ni * 16 + l16;
        float v = acc[mi][ni][r];
        if (C)   C[(size_t)m * N + n] = v;
        if (C16) C16[(size_t)m * N + n] = (_Float16)v;
      }
}

// ---------------------------------------------------------------------------
// LDS-tiled transpose+convert:  W[N,K] f32 (row-major, +rowOff) -> Wt[K,N] f16.
// 32x32 tiles, 256 threads; coalesced global reads (along K) and writes (along N).
// ---------------------------------------------------------------------------
__global__ __launch_bounds__(256) void wt_transpose_kernel(
    const float* __restrict__ W, _Float16* __restrict__ Wt,
    int K, int N, int rowOff)
{
  __shared__ _Float16 tile[32][33];   // +1 pad: no bank conflicts
  const int tx = threadIdx.x & 31;
  const int ty = threadIdx.x >> 5;    // 0..7
  const int nBase = blockIdx.x * 32;
  const int kBase = blockIdx.y * 32;
#pragma unroll
  for (int j = 0; j < 4; ++j) {
    int n = nBase + ty + j * 8;
    tile[ty + j * 8][tx] = (_Float16)W[(size_t)(rowOff + n) * K + kBase + tx];
  }
  __syncthreads();
#pragma unroll
  for (int j = 0; j < 4; ++j) {
    int k = kBase + ty + j * 8;
    Wt[(size_t)k * N + nBase + tx] = tile[tx][ty + j * 8];
  }
}

__global__ void f32_to_f16_kernel(const float* __restrict__ x, _Float16* __restrict__ y, int n)
{
  int i = blockIdx.x * blockDim.x + threadIdx.x;
  if (i < n) y[i] = (_Float16)x[i];
}

// pieces16[b,c] = piece_emb[queue[b,0], c]  (queue is (B,5) int)
__global__ void gather_pieces_kernel(const int* __restrict__ queue,
                                     const float* __restrict__ emb,
                                     _Float16* __restrict__ out)
{
  int i = blockIdx.x * blockDim.x + threadIdx.x;  // B*512
  if (i >= BATCH * DMODEL) return;
  int b = i >> 9, c = i & 511;
  out[i] = (_Float16)emb[queue[b * 5] * DMODEL + c];
}

// 4-key attention per (b, head): scores over r=0..3, softmax, weighted sum of v1.
__global__ void attn4_kernel(const float* __restrict__ q1, const float* __restrict__ k1,
                             const float* __restrict__ v1, _Float16* __restrict__ att16)
{
  int t = blockIdx.x * blockDim.x + threadIdx.x;   // 512*8
  if (t >= BATCH * 8) return;
  int b = t >> 3, n = t & 7;
  const float* q = q1 + (size_t)b * DMODEL + n * 64;
  float sc[4];
#pragma unroll
  for (int r = 0; r < 4; ++r) {
    const float* kp = k1 + (size_t)(b * 4 + r) * DMODEL + n * 64;
    float s = 0.f;
    for (int d = 0; d < 64; ++d) s += q[d] * kp[d];
    sc[r] = s * 0.125f;   // 1/sqrt(64)
  }
  float m = fmaxf(fmaxf(sc[0], sc[1]), fmaxf(sc[2], sc[3]));
  float w[4], sum = 0.f;
#pragma unroll
  for (int r = 0; r < 4; ++r) { w[r] = __expf(sc[r] - m); sum += w[r]; }
  float inv = 1.f / sum;
#pragma unroll
  for (int r = 0; r < 4; ++r) w[r] *= inv;
  for (int d = 0; d < 64; ++d) {
    float o = 0.f;
#pragma unroll
    for (int r = 0; r < 4; ++r) o += w[r] * v1[(size_t)(b * 4 + r) * DMODEL + n * 64 + d];
    att16[(size_t)b * DMODEL + n * 64 + d] = (_Float16)o;
  }
}

// out = LayerNorm(x + y) * g + beta ; one wave32 per 512-wide row.
__global__ __launch_bounds__(32) void add_ln_kernel(
    const float* __restrict__ x, const float* __restrict__ y,
    const float* __restrict__ g, const float* __restrict__ beta,
    float* __restrict__ out, _Float16* __restrict__ out16)
{
  int row = blockIdx.x, lane = threadIdx.x;
  float v[16];
  float s = 0.f;
#pragma unroll
  for (int i = 0; i < 16; ++i) {
    int c = i * 32 + lane;
    v[i] = x[(size_t)row * DMODEL + c] + y[(size_t)row * DMODEL + c];
    s += v[i];
  }
  for (int m = 16; m > 0; m >>= 1) s += __shfl_xor(s, m, 32);
  float mean = s * (1.f / 512.f);
  float vs = 0.f;
#pragma unroll
  for (int i = 0; i < 16; ++i) { float d = v[i] - mean; vs += d * d; }
  for (int m = 16; m > 0; m >>= 1) vs += __shfl_xor(vs, m, 32);
  float rstd = rsqrtf(vs * (1.f / 512.f) + 1e-5f);
#pragma unroll
  for (int i = 0; i < 16; ++i) {
    int c = i * 32 + lane;
    float o = (v[i] - mean) * rstd * g[c] + beta[c];
    out[(size_t)row * DMODEL + c] = o;
    out16[(size_t)row * DMODEL + c] = (_Float16)o;
  }
}

__global__ void add_mem_kernel(const float* __restrict__ y, const float* __restrict__ mem,
                               float* __restrict__ xo, _Float16* __restrict__ xo16, int n)
{
  int i = blockIdx.x * blockDim.x + threadIdx.x;
  if (i >= n) return;
  float v = y[i] + mem[i];
  xo[i] = v;
  xo16[i] = (_Float16)v;
}

// pol head: N=10 (< WMMA tile) -> plain VALU dot products.
__global__ void pol_kernel(const float* __restrict__ xo, const float* __restrict__ W,
                           const float* __restrict__ bsc, float* __restrict__ out)
{
  int t = blockIdx.x * blockDim.x + threadIdx.x;   // 512*10
  if (t >= BATCH * 10) return;
  int b = t / 10, j = t - b * 10;
  const float* xp = xo + (size_t)b * DMODEL;
  const float* wp = W + (size_t)j * DMODEL;
  float s = bsc[j];
  for (int i = 0; i < DMODEL; ++i) s += xp[i] * wp[i];
  out[(size_t)b * 10 + j] = s;
}

// ---------------------------------------------------------------------------
static void launch_gemm(const _Float16* A, const _Float16* Bt, const float* bias,
                        float* C, _Float16* C16, int M, int N, int K, hipStream_t s)
{
  dim3 grid((unsigned)(N / 128), (unsigned)(M / 64));
  wmma_gemm_bias_kernel<<<grid, 128, 0, s>>>(A, Bt, bias, C, C16, M, N, K);
}

static void launch_wt(const float* W, _Float16* Wt, int K, int N, int rowOff, hipStream_t s)
{
  dim3 grid((unsigned)(N / 32), (unsigned)(K / 32));
  wt_transpose_kernel<<<grid, 256, 0, s>>>(W, Wt, K, N, rowOff);
}

extern "C" void kernel_launch(void* const* d_in, const int* in_sizes, int n_in,
                              void* d_out, int out_size, void* d_ws, size_t ws_size,
                              hipStream_t stream)
{
  // Live inputs (dict order): 1 queue, 3 state_in, 4 memory, 9 piece_emb,
  // 10 rot_w, 11 rot_b, 12 q_w, 13 q_b, 14 k_w, 15 k_b, 16 v_w, 17 v_b,
  // 18 qkv_w, 19 qkv_b, 20 queue_w, 21 queue_b, 22 an_g, 23 an_b, 24 fn_g,
  // 25 fn_b, 38 a0_in_w, 39 a0_in_b, 40 a0_out_w, 41 a0_out_b, 42 a1_in_w,
  // 43 a1_in_b, 44 a1_out_w, 45 a1_out_b, 46 pol_w, 47 pol_b, 48 val_w, 49 val_b.
  const int*   queue     = (const int*)  d_in[1];
  const float* state_in  = (const float*)d_in[3];
  const float* memory    = (const float*)d_in[4];
  const float* piece_emb = (const float*)d_in[9];
  const float* rot_w     = (const float*)d_in[10];
  const float* rot_b     = (const float*)d_in[11];
  const float* q_w  = (const float*)d_in[12]; const float* q_b  = (const float*)d_in[13];
  const float* k_w  = (const float*)d_in[14]; const float* k_b  = (const float*)d_in[15];
  const float* v_w  = (const float*)d_in[16]; const float* v_b  = (const float*)d_in[17];
  const float* qkv_w = (const float*)d_in[18]; const float* qkv_b = (const float*)d_in[19];
  const float* queue_w = (const float*)d_in[20]; const float* queue_b = (const float*)d_in[21];
  const float* an_g = (const float*)d_in[22]; const float* an_b = (const float*)d_in[23];
  const float* fn_g = (const float*)d_in[24]; const float* fn_b = (const float*)d_in[25];
  const float* a0_in_w  = (const float*)d_in[38]; const float* a0_in_b  = (const float*)d_in[39];
  const float* a0_out_w = (const float*)d_in[40]; const float* a0_out_b = (const float*)d_in[41];
  const float* a1_in_w  = (const float*)d_in[42]; const float* a1_in_b  = (const float*)d_in[43];
  const float* a1_out_w = (const float*)d_in[44]; const float* a1_out_b = (const float*)d_in[45];
  const float* pol_w = (const float*)d_in[46]; const float* pol_b = (const float*)d_in[47];
  const float* val_w = (const float*)d_in[48]; const float* val_b = (const float*)d_in[49];

  char* ws = (char*)d_ws;
  size_t off = 0;
  auto alloc = [&](size_t bytes) -> char* {
    char* p = ws + off;
    off = (off + bytes + 255) & ~(size_t)255;
    return p;
  };
  const size_t E2 = (size_t)DMODEL * DMODEL;          // 512*512
  // f16 transposed weights (K x N)
  _Float16* wt_rot   = (_Float16*)alloc(512 * 2048 * 2);
  _Float16* wt_a0v   = (_Float16*)alloc(E2 * 2);
  _Float16* wt_a0out = (_Float16*)alloc(E2 * 2);
  _Float16* wt_q     = (_Float16*)alloc(E2 * 2);
  _Float16* wt_k     = (_Float16*)alloc(E2 * 2);
  _Float16* wt_v     = (_Float16*)alloc(E2 * 2);
  _Float16* wt_qkv   = (_Float16*)alloc(E2 * 2);
  _Float16* wt_queue = (_Float16*)alloc(E2 * 2);
  _Float16* wt_a1v   = (_Float16*)alloc(E2 * 2);
  _Float16* wt_a1out = (_Float16*)alloc(E2 * 2);
  _Float16* wt_val   = (_Float16*)alloc(E2 * 2);
  // f16 activations
  _Float16* pieces16 = (_Float16*)alloc(E2 * 2);
  _Float16* state16  = (_Float16*)alloc(E2 * 2);
  _Float16* pwr16    = (_Float16*)alloc(512 * 2048 * 2);
  _Float16* v016     = (_Float16*)alloc(E2 * 2);
  _Float16* s016     = (_Float16*)alloc(E2 * 2);
  _Float16* att16    = (_Float16*)alloc(E2 * 2);
  _Float16* s116     = (_Float16*)alloc(E2 * 2);
  _Float16* s216     = (_Float16*)alloc(E2 * 2);
  _Float16* v216     = (_Float16*)alloc(E2 * 2);
  _Float16* xo16     = (_Float16*)alloc(E2 * 2);
  // f32 activations
  float* s0   = (float*)alloc(E2 * 4);
  float* q1   = (float*)alloc(E2 * 4);
  float* k1   = (float*)alloc(2048 * 512 * 4);
  float* v1   = (float*)alloc(2048 * 512 * 4);
  float* yatt = (float*)alloc(E2 * 4);
  float* s1   = (float*)alloc(E2 * 4);
  float* yq   = (float*)alloc(E2 * 4);
  float* s2   = (float*)alloc(E2 * 4);
  float* y1   = (float*)alloc(E2 * 4);
  float* xo   = (float*)alloc(E2 * 4);

  // --- weight transpose + f16 convert (LDS-tiled, fully coalesced) ---
  launch_wt(rot_w,    wt_rot,   512, 2048, 0,    stream);
  launch_wt(a0_in_w,  wt_a0v,   512, 512,  1024, stream);  // v-slice of a0_in_w
  launch_wt(a0_out_w, wt_a0out, 512, 512,  0,    stream);
  launch_wt(q_w,      wt_q,     512, 512,  0,    stream);
  launch_wt(k_w,      wt_k,     512, 512,  0,    stream);
  launch_wt(v_w,      wt_v,     512, 512,  0,    stream);
  launch_wt(qkv_w,    wt_qkv,   512, 512,  0,    stream);
  launch_wt(queue_w,  wt_queue, 512, 512,  0,    stream);
  launch_wt(a1_in_w,  wt_a1v,   512, 512,  1024, stream);  // v-slice of a1_in_w
  launch_wt(a1_out_w, wt_a1out, 512, 512,  0,    stream);
  launch_wt(val_w,    wt_val,   512, 512,  0,    stream);

  // --- activations ---
  gather_pieces_kernel<<<cdivu(BATCH * DMODEL, 256), 256, 0, stream>>>(queue, piece_emb, pieces16);
  f32_to_f16_kernel<<<cdivu(BATCH * DMODEL, 256), 256, 0, stream>>>(state_in, state16, BATCH * DMODEL);

  // pwr = pieces @ rot_w.T + rot_b  (512 x 2048), consumed only as f16
  launch_gemm(pieces16, wt_rot, rot_b, nullptr, pwr16, 512, 2048, 512, stream);

  // a0: only v-projection + out-projection survive (Lk==1 => softmax == 1)
  launch_gemm(state16, wt_a0v, a0_in_b + 1024, nullptr, v016, 512, 512, 512, stream);
  launch_gemm(v016, wt_a0out, a0_out_b, s0, s016, 512, 512, 512, stream);

  // second attention over 4 rotation keys (the only non-trivial softmax)
  launch_gemm(s016,  wt_q, q_b, q1, nullptr, 512,  512, 512, stream);
  launch_gemm(pwr16, wt_k, k_b, k1, nullptr, 2048, 512, 512, stream);   // (B*4) x 512
  launch_gemm(pwr16, wt_v, v_b, v1, nullptr, 2048, 512, 512, stream);
  attn4_kernel<<<cdivu(BATCH * 8, 128), 128, 0, stream>>>(q1, k1, v1, att16);

  // s1 = LN(s0 + att @ qkv_w.T + qkv_b)
  launch_gemm(att16, wt_qkv, qkv_b, yatt, nullptr, 512, 512, 512, stream);
  add_ln_kernel<<<BATCH, 32, 0, stream>>>(s0, yatt, an_g, an_b, s1, s116);

  // s2 = LN(s1 + s1 @ queue_w.T + queue_b)
  launch_gemm(s116, wt_queue, queue_b, yq, nullptr, 512, 512, 512, stream);
  add_ln_kernel<<<BATCH, 32, 0, stream>>>(s1, yq, fn_g, fn_b, s2, s216);

  // a1: uniform softmax over 200 identical keys => out == v2
  launch_gemm(s216, wt_a1v, a1_in_b + 1024, nullptr, v216, 512, 512, 512, stream);
  launch_gemm(v216, wt_a1out, a1_out_b, y1, nullptr, 512, 512, 512, stream);

  // xo = y1 + memory
  add_mem_kernel<<<cdivu(BATCH * DMODEL, 256), 256, 0, stream>>>(y1, memory, xo, xo16, BATCH * DMODEL);

  // outputs: pol (B,10) first, then val (B,512)
  float* out_pol = (float*)d_out;
  float* out_val = out_pol + (size_t)BATCH * 10;
  launch_gemm(xo16, wt_val, val_b, out_val, nullptr, 512, 512, 512, stream);
  pol_kernel<<<cdivu(BATCH * 10, 128), 128, 0, stream>>>(xo, pol_w, pol_b, out_pol);
}
